// EquivariantMixBlock_46205258170438
// MI455X (gfx1250) — compile-verified, matrix operand order checked by reading, not run on hardware
//
#include <hip/hip_runtime.h>

// ---------------------------------------------------------------------------
// EquivariantMixBlock for MI455X (gfx1250, wave32, WMMA + TDM).
//
// Dominant cost: per-edge radial MLP  silu(len*w1+b1) @ w2  (1x64 @ 64x576),
// batched 16 edges/wave -> 16x64 @ 64x576 GEMM via v_wmma_f32_16x16x32_f16.
// Shared w2 (f16, K-contiguous) is staged into LDS once per block with the
// Tensor Data Mover (pad_enable gives a bank-padded 144B row stride).
// Equivariant contractions run as uniform (divergence-free) wave passes.
// ---------------------------------------------------------------------------

typedef __attribute__((ext_vector_type(16))) _Float16 v16h;
typedef __attribute__((ext_vector_type(8)))  float    v8f;
typedef unsigned int uint32x4 __attribute__((ext_vector_type(4)));
typedef int          int32x8  __attribute__((ext_vector_type(8)));
typedef int          int32x4  __attribute__((ext_vector_type(4)));

#define MUL0 16
#define MUL1 8
#define DIM  40          // MUL0 + 3*MUL1
#define RMLP 64
#define WNUM 576         // 256 + 128 + 128 + 64
#define NTILES 36        // 576 / 16

#define EDGES_PER_WAVE  16
#define WAVES_PER_BLOCK 4
#define EDGES_PER_BLOCK (EDGES_PER_WAVE * WAVES_PER_BLOCK)

// ---- block-shared LDS: w2 staged as [576 cols][64 halfs + 8 pad] ----------
#define WS_ROW_HALFS 72                                 // 144B rows (16B-mult, bank-padded)
#define WS_BYTES     (WNUM * WS_ROW_HALFS * 2)          // 82944

// ---- per-wave LDS (wave-private -> no cross-wave barriers needed) ---------
#define ACT_STRIDE 72                                   // halfs/row, 144B
#define ACT_BYTES  (EDGES_PER_WAVE * ACT_STRIDE * 2)    // 2304
#define W_STRIDE   580                                  // floats/row (576 + pad)
#define W_BYTES    (EDGES_PER_WAVE * W_STRIDE * 4)      // 37120
#define XS_OFF     0                                    // [24] = hs(16) ++ inv_sqrt3*dot(8)
#define HV_OFF     24                                   // [24] = hv (8x3)
#define SH_OFF     48                                   // [3] sh1 + [1] recv (int)
#define SCR_STRIDE 52
#define SCR_BYTES  (EDGES_PER_WAVE * SCR_STRIDE * 4)    // 3328
#define WAVE_LDS   (ACT_BYTES + W_BYTES + SCR_BYTES)    // 42752
#define BLOCK_LDS  (WS_BYTES + WAVE_LDS * WAVES_PER_BLOCK)   // 253952 (<=320KB/WGP)

#define C_N0        0.2041241452319315f   // sqrt(1/24)
#define C_N1        0.3535533905932738f   // sqrt(3/24)
#define C_SQRT3     1.7320508075688772f
#define C_INV_SQRT3 0.5773502691896258f

__device__ __forceinline__ float silu_f(float x) {
  return x / (1.0f + __expf(-x));
}

// ---- prep: w2 (f32 [64][576]) -> f16 transposed [576][64] (K-contiguous) ----
__global__ void eqmix_prep_w2(const float* __restrict__ w2, _Float16* __restrict__ w2t) {
  int t = blockIdx.x * blockDim.x + threadIdx.x;
  if (t >= WNUM * RMLP) return;
  int n = t >> 6;        // output column 0..575
  int k = t & 63;        // hidden index 0..63
  w2t[n * RMLP + k] = (_Float16)w2[k * WNUM + n];
}

__global__ void eqmix_zero(float* __restrict__ p, int n) {
  int i = blockIdx.x * blockDim.x + threadIdx.x;
  if (i < n) p[i] = 0.0f;
}

// ---------------------------- main edge kernel -----------------------------
__global__ __launch_bounds__(WAVES_PER_BLOCK * 32)
void eqmix_edge_kernel(const float* __restrict__ h,
                       const int*   __restrict__ edge_index,   // [2][E]
                       const float* __restrict__ edge_vec,     // [E][3]
                       const float* __restrict__ edge_len,     // [E]
                       const float* __restrict__ mlp_w1,       // [64]
                       const float* __restrict__ mlp_b1,       // [64]
                       const float* __restrict__ mlp_b2,       // [576]
                       const _Float16* __restrict__ w2t,       // [576][64] f16
                       float* __restrict__ agg,                // [N][40]
                       int E)
{
  extern __shared__ char smem[];
  const int lane = threadIdx.x & 31;
  const int wave = threadIdx.x >> 5;
  const int m    = lane & 15;     // tile row (A:M) / tile col (B,C,D:N)
  const int g    = lane >> 4;     // lane group -> K-range selector

  _Float16* wsB = (_Float16*)smem;                        // staged weights
  char*     waveLds = smem + WS_BYTES + wave * WAVE_LDS;
  _Float16* actL = (_Float16*)waveLds;
  float*    wL   = (float*)(waveLds + ACT_BYTES);
  float*    scrL = (float*)(waveLds + ACT_BYTES + W_BYTES);

  const int tileBase = blockIdx.x * EDGES_PER_BLOCK + wave * EDGES_PER_WAVE;

  // ---- Weight staging: TDM async copy global->LDS, 16B pad every 128B row.
#if defined(__has_builtin) && __has_builtin(__builtin_amdgcn_tensor_load_to_lds)
  #define EQMIX_USE_TDM 1
#else
  #define EQMIX_USE_TDM 0
#endif

#if EQMIX_USE_TDM
  if (wave == 0) {
    const unsigned ldsAddr = (unsigned)(uintptr_t)(void*)wsB;
    const unsigned long long ga = (unsigned long long)(uintptr_t)(const void*)w2t;
    const unsigned nElem = (WNUM * RMLP * 2) / 8;         // 9216 x 8B elements
    // D# group0: count=1 | lds_addr | global_addr | type=2
    uint32x4 g0 = { 1u, ldsAddr, (unsigned)ga,
                    (unsigned)((ga >> 32) & 0x01FFFFFFu) | 0x80000000u };
    // D# group1: data_size=8B, pad_enable, pad every 128B (code 4) by 16B (code 3)
    int32x8 g1 = { (int)((3u << 16) | (1u << 20) | (4u << 22) | (3u << 25)),
                   (int)((nElem & 0xFFFFu) << 16),        // tensor_dim0[15:0]
                   (int)(((nElem >> 16) & 0xFFFFu) | (1u << 16)),  // dim0 hi | tensor_dim1=1
                   (int)((nElem & 0xFFFFu) << 16),        // tile_dim0 = nElem
                   (int)1,                                // tile_dim1=1, tile_dim2=0
                   (int)nElem,                            // tensor_dim0_stride lo
                   0, 0 };
    int32x4 gz = { 0, 0, 0, 0 };
#if __clang_major__ >= 23
    int32x8 gz8 = { 0, 0, 0, 0, 0, 0, 0, 0 };
    __builtin_amdgcn_tensor_load_to_lds(g0, g1, gz, gz, gz8, 0);
#else
    __builtin_amdgcn_tensor_load_to_lds(g0, g1, gz, gz, 0);
#endif
  }
#else
  // Fallback: cooperative padded copy (36 float4 per thread)
  for (int q = threadIdx.x; q < (WNUM * RMLP * 2) / 16; q += blockDim.x) {
    float4 v = *(const float4*)((const char*)w2t + q * 16);
    *(float4*)((char*)wsB + (q >> 3) * 144 + (q & 7) * 16) = v;
  }
#endif

  // ---- Stage 0 (overlaps TDM): act[e][r] = silu(len*w1[r]+b1[r]) -> LDS f16
  {
    const int  e     = m;
    const int  eg    = tileBase + e;
    const bool valid = (eg < E);
    const float len  = valid ? edge_len[eg] : 0.0f;
    const int  r0    = g * 32;
#pragma unroll
    for (int rr = 0; rr < 32; ++rr) {
      const int r = r0 + rr;
      float a = silu_f(len * mlp_w1[r] + mlp_b1[r]);
      if (!valid) a = 0.0f;
      actL[e * ACT_STRIDE + r] = (_Float16)a;
    }
  }
  asm volatile("s_wait_dscnt 0" ::: "memory");   // act store -> load (same wave)

#if EQMIX_USE_TDM
  if (wave == 0) {
#if __has_builtin(__builtin_amdgcn_s_wait_tensorcnt)
    __builtin_amdgcn_s_wait_tensorcnt(0);
#else
    asm volatile("s_wait_tensorcnt 0" ::: "memory");
#endif
  }
#endif
  __syncthreads();                               // weights visible to all waves

  // ---- Stage 1: W[16 edges][576] = act(16x64) @ w2(64x576) + b2 via WMMA ----
  // f16 A/B fragment layout: halfs 0..7 -> K = 8g.. , halfs 8..15 -> K = 16+8g..
  v16h a0, a1;
  {
    const _Float16* row = actL + m * ACT_STRIDE;
    ((float4*)&a0)[0] = *(const float4*)(row +      8 * g);
    ((float4*)&a0)[1] = *(const float4*)(row + 16 + 8 * g);
    ((float4*)&a1)[0] = *(const float4*)(row + 32 + 8 * g);
    ((float4*)&a1)[1] = *(const float4*)(row + 48 + 8 * g);
  }

  for (int nt = 0; nt < NTILES; ++nt) {
    const int col = nt * 16 + m;                       // B: lane holds column N
    const _Float16* bcol = wsB + col * WS_ROW_HALFS;   // LDS, 144B row stride
    v16h b0, b1;
    ((float4*)&b0)[0] = *(const float4*)(bcol +      8 * g);
    ((float4*)&b0)[1] = *(const float4*)(bcol + 16 + 8 * g);
    ((float4*)&b1)[0] = *(const float4*)(bcol + 32 + 8 * g);
    ((float4*)&b1)[1] = *(const float4*)(bcol + 48 + 8 * g);

    const float bias = mlp_b2[col];                    // same N for this lane's 8 rows
    v8f c = {bias, bias, bias, bias, bias, bias, bias, bias};
    c = __builtin_amdgcn_wmma_f32_16x16x32_f16(false, a0, false, b0, (short)0, c, false, false);
    c = __builtin_amdgcn_wmma_f32_16x16x32_f16(false, a1, false, b1, (short)0, c, false, false);
    // D layout: lane -> N = lane%16, reg r -> M = r + 8*(lane/16)
#pragma unroll
    for (int r = 0; r < 8; ++r)
      wL[(r + 8 * g) * W_STRIDE + nt * 16 + m] = c[r];
  }

  // ---- Stage 2 prep: gather sender features, sh1 / dot / recv -> LDS scratch
  if (lane < 16) {
    const int e      = lane;
    const int eg     = tileBase + e;
    const int eclamp = (eg < E) ? eg : (E - 1);
    const float vx = edge_vec[eclamp * 3 + 0];
    const float vy = edge_vec[eclamp * 3 + 1];
    const float vz = edge_vec[eclamp * 3 + 2];
    const float inv = rsqrtf(vx * vx + vy * vy + vz * vz + 1e-30f);
    const float s0 = C_SQRT3 * vx * inv;
    const float s1 = C_SQRT3 * vy * inv;
    const float s2 = C_SQRT3 * vz * inv;
    const int snd = edge_index[eclamp];                 // sender row
    const float4* hp = (const float4*)(h + (size_t)snd * DIM);   // 160B row
    float hg[DIM];
#pragma unroll
    for (int q = 0; q < 10; ++q) {
      float4 t = hp[q];
      hg[q * 4 + 0] = t.x; hg[q * 4 + 1] = t.y; hg[q * 4 + 2] = t.z; hg[q * 4 + 3] = t.w;
    }
    float* xs = scrL + e * SCR_STRIDE + XS_OFF;
    float* hv = scrL + e * SCR_STRIDE + HV_OFF;
    float* sh = scrL + e * SCR_STRIDE + SH_OFF;
#pragma unroll
    for (int u = 0; u < 16; ++u) xs[u] = hg[u];
#pragma unroll
    for (int u = 0; u < 8; ++u) {
      const float hx = hg[16 + u * 3 + 0];
      const float hy = hg[16 + u * 3 + 1];
      const float hz = hg[16 + u * 3 + 2];
      xs[16 + u] = C_INV_SQRT3 * (hx * s0 + hy * s1 + hz * s2);   // inv_sqrt3 * dot
      hv[u * 3 + 0] = hx; hv[u * 3 + 1] = hy; hv[u * 3 + 2] = hz;
    }
    sh[0] = s0; sh[1] = s1; sh[2] = s2;
    ((int*)sh)[3] = (eg < E) ? edge_index[E + eg] : 0;  // receiver cached in LDS
  }
  asm volatile("s_wait_dscnt 0" ::: "memory");

  // ---- Stage 2: uniform (divergence-free) message passes + scatter-add ----
  // W column map: W1=[0,256) u*16+j ; W2=[256,384) u*16+j ;
  //               W3=[384,512) u*8+w ; W4=[512,576) u*8+w.

  // scalar outputs: 16 edges x 16 comps = 8 full-wave passes
#pragma unroll
  for (int p = 0; p < 8; ++p) {
    const int idx = p * 32 + lane;
    const int e = idx >> 4;
    const int j = idx & 15;
    const float* W  = wL + e * W_STRIDE;
    const float* xs = scrL + e * SCR_STRIDE + XS_OFF;
    float s = 0.0f;
#pragma unroll
    for (int u = 0; u < 16; ++u) s += xs[u] * W[u * 16 + j];
#pragma unroll
    for (int u = 0; u < 8; ++u)  s += xs[16 + u] * W[256 + u * 16 + j];
    const int recv = ((const int*)(scrL + e * SCR_STRIDE + SH_OFF))[3];
    if (tileBase + e < E)
      atomicAdd(agg + (size_t)recv * DIM + j, C_N0 * s);
  }

  // vector outputs: 16 edges x 24 comps = 12 full-wave passes
#pragma unroll
  for (int p = 0; p < 12; ++p) {
    const int idx = p * 32 + lane;
    const int e = idx / 24;
    const int t = idx - e * 24;
    const int w = t / 3;
    const int k = t - w * 3;
    const float* W  = wL + e * W_STRIDE;
    const float* xs = scrL + e * SCR_STRIDE + XS_OFF;
    const float* hv = scrL + e * SCR_STRIDE + HV_OFF;
    const float* sh = scrL + e * SCR_STRIDE + SH_OFF;
    float t3 = 0.0f, t4 = 0.0f;
#pragma unroll
    for (int u = 0; u < 16; ++u) t3 += xs[u] * W[384 + u * 8 + w];
#pragma unroll
    for (int u = 0; u < 8; ++u)  t4 += hv[u * 3 + k] * W[512 + u * 8 + w];
    const float acc = (C_N1 * C_INV_SQRT3) * (t3 * sh[k] + t4);
    const int recv = ((const int*)sh)[3];
    if (tileBase + e < E)
      atomicAdd(agg + (size_t)recv * DIM + 16 + t, acc);
  }
}

// ---- finalize: gate = sigmoid(h[:,:16] @ gate_w + gate_b); out = h + gated agg
__global__ void eqmix_finalize(const float* __restrict__ h, const float* __restrict__ agg,
                               const float* __restrict__ gate_w, const float* __restrict__ gate_b,
                               float* __restrict__ out, int n)
{
  int i = blockIdx.x * blockDim.x + threadIdx.x;
  if (i >= n) return;
  const float* hi = h + (size_t)i * DIM;
  const float* ai = agg + (size_t)i * DIM;
  float* oi = out + (size_t)i * DIM;
  float hs[16];
#pragma unroll
  for (int u = 0; u < 16; ++u) { hs[u] = hi[u]; oi[u] = hs[u] + ai[u]; }
#pragma unroll
  for (int mo = 0; mo < 24; ++mo) {
    float z = gate_b[mo];
#pragma unroll
    for (int u = 0; u < 16; ++u) z += hs[u] * gate_w[u * 24 + mo];
    const float gsig = 1.0f / (1.0f + __expf(-z));
    oi[16 + mo] = hi[16 + mo] + ai[16 + mo] * gsig;
  }
}

// ---------------------------------------------------------------------------
extern "C" void kernel_launch(void* const* d_in, const int* in_sizes, int n_in,
                              void* d_out, int out_size, void* d_ws, size_t ws_size,
                              hipStream_t stream)
{
  const float* h    = (const float*)d_in[0];
  const int*   eidx = (const int*)  d_in[1];
  const float* evec = (const float*)d_in[2];
  const float* elen = (const float*)d_in[3];
  const float* w1   = (const float*)d_in[4];
  const float* b1   = (const float*)d_in[5];
  const float* w2   = (const float*)d_in[6];
  const float* b2   = (const float*)d_in[7];
  const float* gw   = (const float*)d_in[8];
  const float* gb   = (const float*)d_in[9];
  float* out = (float*)d_out;

  const int E = in_sizes[3];            // edge_len count
  const int N = in_sizes[0] / DIM;      // h rows

  // workspace: agg[N][40] f32, then transposed f16 weights
  float* agg = (float*)d_ws;
  size_t aggBytes = ((size_t)N * DIM * sizeof(float) + 255) & ~(size_t)255;
  _Float16* w2t = (_Float16*)((char*)d_ws + aggBytes);

  const int ztot = N * DIM;
  eqmix_zero<<<(ztot + 255) / 256, 256, 0, stream>>>(agg, ztot);
  eqmix_prep_w2<<<(WNUM * RMLP + 255) / 256, 256, 0, stream>>>(w2, w2t);

  const int blocks = (E + EDGES_PER_BLOCK - 1) / EDGES_PER_BLOCK;
  eqmix_edge_kernel<<<blocks, WAVES_PER_BLOCK * 32, BLOCK_LDS, stream>>>(
      h, eidx, evec, elen, w1, b1, b2, w2t, agg, E);

  eqmix_finalize<<<(N + 255) / 256, 256, 0, stream>>>(h, agg, gw, gb, out, N);
}
